// LiftingAttention_4389456576535
// MI455X (gfx1250) — matrix operand order repaired and checked by reading, not compiled, not run on hardware
//
#include <hip/hip_runtime.h>

typedef __bf16 bf16;
typedef __attribute__((ext_vector_type(16))) __bf16 v16bf;
typedef __attribute__((ext_vector_type(8)))  float  v8f;

#define DD  512
#define HH  8
#define GG  8
#define BBb 4
#define NNN 512
#define DKK 64

static __device__ __forceinline__ v8f wmma_bf16(v16bf a, v16bf b, v8f c) {
  // (neg_a, A, neg_b, B, c_mod, C, reuse_a, reuse_b)
  return __builtin_amdgcn_wmma_f32_16x16x32_bf16(false, a, false, b, (short)0, c, false, false);
}

// Load a 16x32 bf16 fragment half: two contiguous 16B chunks per lane.
static __device__ __forceinline__ v16bf ld_frag32(const bf16* p0, const bf16* p1) {
  v16bf f;
  ((uint4*)&f)[0] = *(const uint4*)p0;
  ((uint4*)&f)[1] = *(const uint4*)p1;
  return f;
}

// ---------------- weight convert: Wt[c][k] = (bf16)W[k][c] ----------------
__global__ void wconv_kernel(const float* __restrict__ W, bf16* __restrict__ Wt) {
  int idx = blockIdx.x * 256 + threadIdx.x;   // 262144 elements
  int c = idx >> 9;
  int k = idx & 511;
  Wt[idx] = (bf16)W[(size_t)k * DD + c];
}

// ---------------- LayerNorm: one block per row ----------------
__global__ void ln_kernel(const float* __restrict__ X, const float* __restrict__ g,
                          const float* __restrict__ b, bf16* __restrict__ Xn) {
  const int row  = blockIdx.x;              // B*N rows
  const int tid  = threadIdx.x;             // 256 threads, 2 elems each
  const int wave = tid >> 5, lane = tid & 31;
  const float* x = X + (size_t)row * DD;
  float v0 = x[tid], v1 = x[tid + 256];

  __shared__ float red[8];
  __shared__ float mu_s, rs_s;

  float s = v0 + v1;
#pragma unroll
  for (int d = 16; d >= 1; d >>= 1) s += __shfl_xor(s, d, 32);
  if (lane == 0) red[wave] = s;
  __syncthreads();
  if (tid == 0) {
    float t = 0.f;
#pragma unroll
    for (int i = 0; i < 8; ++i) t += red[i];
    mu_s = t * (1.0f / DD);
  }
  __syncthreads();
  const float mu = mu_s;
  const float d0 = v0 - mu, d1 = v1 - mu;
  float vs = d0 * d0 + d1 * d1;
#pragma unroll
  for (int d = 16; d >= 1; d >>= 1) vs += __shfl_xor(vs, d, 32);
  __syncthreads();
  if (lane == 0) red[wave] = vs;
  __syncthreads();
  if (tid == 0) {
    float t = 0.f;
#pragma unroll
    for (int i = 0; i < 8; ++i) t += red[i];
    rs_s = rsqrtf(t * (1.0f / DD) + 1e-5f);
  }
  __syncthreads();
  const float rs = rs_s;
  Xn[(size_t)row * DD + tid]       = (bf16)(d0 * rs * g[tid]       + b[tid]);
  Xn[(size_t)row * DD + tid + 256] = (bf16)(d1 * rs * g[tid + 256] + b[tid + 256]);
}

// ---------------- WMMA GEMM (software pipelined) ----------------
// C(Mx512) = A(Mx512,bf16) * Wt^T + bias
// mode 0: bf16 out, layout [b][h][n][dk]   (Q, K)
// mode 2: bf16 out, layout [b][h][dk][n]   (V, transposed for A*V B-fragments)
// mode 1: f32  out, row-major Mx512        (final projection -> d_out)
__global__ void gemm_kernel(const bf16* __restrict__ A, const bf16* __restrict__ Wt,
                            const float* __restrict__ bias, void* __restrict__ outp,
                            int M, int mode) {
  (void)M;
  const int lane = threadIdx.x & 31;
  const int wave = threadIdx.x >> 5;      // 8 waves: 128-row block tile
  const int bm = blockIdx.x >> 3;
  const int bn = blockIdx.x & 7;          // 8 column tiles of 64
  const int rowBase = bm * 128 + wave * 16;
  const int colBase = bn * 64;
  const int m = lane & 15, half = lane >> 4;

  const bf16* Arow  = A + (size_t)(rowBase + m) * DD;
  const bf16* Wcol0 = Wt + (size_t)(colBase + m) * DD + half * 8;

  // prologue: fragments for k0 = 0
  v16bf af = ld_frag32(Arow + half * 8, Arow + 16 + half * 8);
  v16bf bfrag[4];
#pragma unroll
  for (int j = 0; j < 4; ++j)
    bfrag[j] = ld_frag32(Wcol0 + (size_t)j * 16 * DD, Wcol0 + (size_t)j * 16 * DD + 16);

  v8f acc[4] = {};
#pragma unroll
  for (int k0 = 0; k0 < DD - 32; k0 += 32) {
    const int kn = k0 + 32;
    __builtin_prefetch(Arow + kn + 32, 0, 1);
    // issue next-iteration loads before this iteration's WMMAs
    v16bf afn = ld_frag32(Arow + kn + half * 8, Arow + kn + 16 + half * 8);
    v16bf bn[4];
#pragma unroll
    for (int j = 0; j < 4; ++j) {
      const bf16* w = Wcol0 + (size_t)j * 16 * DD + kn;
      bn[j] = ld_frag32(w, w + 16);
    }
#pragma unroll
    for (int j = 0; j < 4; ++j) acc[j] = wmma_bf16(af, bfrag[j], acc[j]);
    af = afn;
#pragma unroll
    for (int j = 0; j < 4; ++j) bfrag[j] = bn[j];
  }
#pragma unroll
  for (int j = 0; j < 4; ++j) acc[j] = wmma_bf16(af, bfrag[j], acc[j]);  // epilogue

  float bj[4];
#pragma unroll
  for (int j = 0; j < 4; ++j) bj[j] = bias[colBase + j * 16 + m];

  if (mode == 1) {
    float* out = (float*)outp;
#pragma unroll
    for (int j = 0; j < 4; ++j)
#pragma unroll
      for (int r = 0; r < 8; ++r) {
        int row = rowBase + r + 8 * half;
        int col = colBase + j * 16 + m;
        out[(size_t)row * DD + col] = acc[j][r] + bj[j];
      }
  } else {
    bf16* out = (bf16*)outp;
#pragma unroll
    for (int j = 0; j < 4; ++j)
#pragma unroll
      for (int r = 0; r < 8; ++r) {
        int row = rowBase + r + 8 * half;      // = b*512 + n
        int col = colBase + j * 16 + m;        // = h*64 + dk
        int bb = row >> 9, n = row & 511;
        int h = col >> 6, dk = col & 63;
        size_t addr = (mode == 0)
            ? ((((size_t)bb * HH + h) * NNN + n) * DKK + dk)
            : ((((size_t)bb * HH + h) * DKK + dk) * NNN + n);
        out[addr] = (bf16)(acc[j][r] + bj[j]);
      }
  }
}

// ---------------- flash attention with lifted group bias ----------------
// Q,K: [b][h][n][dk] bf16 ; Vt: [b][h][dk][n] bf16 ; O: [b][n][g][h*64+dk] bf16
__global__ void attn_kernel(const bf16* __restrict__ Qm, const bf16* __restrict__ Km,
                            const bf16* __restrict__ Vt, const int* __restrict__ sdiff,
                            const int* __restrict__ gact, const float* __restrict__ rho,
                            bf16* __restrict__ O) {
  const int t  = blockIdx.x;               // 1024 blocks
  const int qb = t & 3, g = (t >> 2) & 7, h = (t >> 5) & 7, b = t >> 8;

  __shared__ int   ga_s[NNN];
  __shared__ float rho_s[NNN];
  __shared__ __align__(16) bf16 pbuf[8][16][40];   // per-wave P transpose staging

  for (int i = threadIdx.x; i < NNN; i += 256) {
    ga_s[i]  = gact[g * NNN + i];
    rho_s[i] = rho[i * HH + h];
  }
  __syncthreads();

  const int wave = threadIdx.x >> 5, lane = threadIdx.x & 31;
  const int m = lane & 15, half = lane >> 4;
  const int qt = qb * 8 + wave;            // q-tile 0..31

  const bf16* Qbase = Qm + (((size_t)(b * HH + h) * NNN) + qt * 16 + m) * DKK;
  v16bf qf0 = ld_frag32(Qbase + half * 8,      Qbase + 16 + half * 8);
  v16bf qf1 = ld_frag32(Qbase + 32 + half * 8, Qbase + 48 + half * 8);

  const bf16* Kbase = Km + ((size_t)(b * HH + h) * NNN) * DKK + half * 8;
  const bf16* Vbase = Vt + ((size_t)(b * HH + h) * DKK) * NNN + half * 8;

  // K fragments for step kt: frag pairs at key rows kt*32+m and kt*32+16+m
  auto ldKfrags = [&](int kt, v16bf kf[4]) {
    const bf16* kp0 = Kbase + (size_t)(kt * 32 + m) * DKK;
    kf[0] = ld_frag32(kp0,      kp0 + 16);      // dk 0..31
    kf[1] = ld_frag32(kp0 + 32, kp0 + 48);      // dk 32..63
    const bf16* kp1 = Kbase + (size_t)(kt * 32 + 16 + m) * DKK;
    kf[2] = ld_frag32(kp1,      kp1 + 16);
    kf[3] = ld_frag32(kp1 + 32, kp1 + 48);
  };

  v8f oacc[4] = {};
  float mrow[8], lrow[8];
#pragma unroll
  for (int r = 0; r < 8; ++r) { mrow[r] = -1e30f; lrow[r] = 0.f; }

  v16bf kf[4];
  ldKfrags(0, kf);

  for (int kt = 0; kt < NNN / 32; ++kt) {   // 32 keys per step
    // V fragments for this step: independent of softmax -> issue first
    v16bf vf[4];
#pragma unroll
    for (int j = 0; j < 4; ++j) {
      const bf16* vp = Vbase + (size_t)(j * 16 + m) * NNN + kt * 32;
      vf[j] = ld_frag32(vp, vp + 16);
    }

    v8f s0 = {}, s1 = {};
    s0 = wmma_bf16(qf0, kf[0], s0);
    s0 = wmma_bf16(qf1, kf[1], s0);
    s1 = wmma_bf16(qf0, kf[2], s1);
    s1 = wmma_bf16(qf1, kf[3], s1);

    // prefetch next step's K fragments; overlaps the softmax VALU work
    v16bf kn[4];
    ldKfrags((kt + 1) & 15, kn);

    const int k0c = kt * 32 + m;
    const int k1c = k0c + 16;
    float p0[8], p1[8];
#pragma unroll
    for (int r = 0; r < 8; ++r) {
      int q = qt * 16 + r + 8 * half;
      float b0 = rho_s[ga_s[sdiff[q * NNN + k0c]]];
      float b1 = rho_s[ga_s[sdiff[q * NNN + k1c]]];
      p0[r] = s0[r] * 0.125f + b0;        // 1/sqrt(DK)=1/8
      p1[r] = s1[r] * 0.125f + b1;
    }
#pragma unroll
    for (int r = 0; r < 8; ++r) {
      float tmax = fmaxf(p0[r], p1[r]);
#pragma unroll
      for (int d = 1; d < 16; d <<= 1) tmax = fmaxf(tmax, __shfl_xor(tmax, d, 32));
      float nm    = fmaxf(mrow[r], tmax);
      float alpha = __expf(mrow[r] - nm);
      p0[r] = __expf(p0[r] - nm);
      p1[r] = __expf(p1[r] - nm);
      float rs = p0[r] + p1[r];
#pragma unroll
      for (int d = 1; d < 16; d <<= 1) rs += __shfl_xor(rs, d, 32);
      lrow[r] = lrow[r] * alpha + rs;
      mrow[r] = nm;
#pragma unroll
      for (int j = 0; j < 4; ++j) oacc[j][r] *= alpha;
    }
    // transpose P (C-layout) -> A-fragment layout via per-wave LDS
#pragma unroll
    for (int r = 0; r < 8; ++r) {
      int row = r + 8 * half;
      pbuf[wave][row][m]      = (bf16)p0[r];
      pbuf[wave][row][16 + m] = (bf16)p1[r];
    }
    asm volatile("s_wait_dscnt 0" ::: "memory");
    const bf16* pp = &pbuf[wave][m][0];
    v16bf af = ld_frag32(pp + half * 8, pp + 16 + half * 8);
#pragma unroll
    for (int j = 0; j < 4; ++j) oacc[j] = wmma_bf16(af, vf[j], oacc[j]);

#pragma unroll
    for (int j = 0; j < 4; ++j) kf[j] = kn[j];
  }
#pragma unroll
  for (int j = 0; j < 4; ++j)
#pragma unroll
    for (int r = 0; r < 8; ++r) {
      int q  = qt * 16 + r + 8 * half;
      int dk = j * 16 + m;
      size_t orow = (size_t)(b * NNN + q) * GG + g;
      O[orow * DD + h * DKK + dk] = (bf16)(oacc[j][r] / lrow[r]);
    }
}

extern "C" void kernel_launch(void* const* d_in, const int* in_sizes, int n_in,
                              void* d_out, int out_size, void* d_ws, size_t ws_size,
                              hipStream_t stream) {
  (void)in_sizes; (void)n_in; (void)out_size; (void)ws_size;
  const float* X   = (const float*)d_in[0];
  const int*   sd  = (const int*)d_in[1];
  const int*   ga  = (const int*)d_in[2];
  const float* lng = (const float*)d_in[3];
  const float* lnb = (const float*)d_in[4];
  const float* Wq  = (const float*)d_in[5];
  const float* bq  = (const float*)d_in[6];
  const float* Wk  = (const float*)d_in[7];
  const float* bk  = (const float*)d_in[8];
  const float* Wv  = (const float*)d_in[9];
  const float* bv  = (const float*)d_in[10];
  const float* Wo  = (const float*)d_in[11];
  const float* bo  = (const float*)d_in[12];
  const float* rho = (const float*)d_in[13];

  char* ws = (char*)d_ws;
  const size_t WSZ = (size_t)DD * DD * sizeof(bf16);        // 512 KB
  const size_t XSZ = (size_t)BBb * NNN * DD * sizeof(bf16); // 2 MB
  bf16* WqT = (bf16*)(ws + 0 * WSZ);
  bf16* WkT = (bf16*)(ws + 1 * WSZ);
  bf16* WvT = (bf16*)(ws + 2 * WSZ);
  bf16* WoT = (bf16*)(ws + 3 * WSZ);
  bf16* Xn  = (bf16*)(ws + 4 * WSZ);
  bf16* Qb  = (bf16*)(ws + 4 * WSZ + 1 * XSZ);
  bf16* Kb  = (bf16*)(ws + 4 * WSZ + 2 * XSZ);
  bf16* Vb  = (bf16*)(ws + 4 * WSZ + 3 * XSZ);
  bf16* Ob  = (bf16*)(ws + 4 * WSZ + 4 * XSZ);              // 16 MB; total 26 MB

  wconv_kernel<<<1024, 256, 0, stream>>>(Wq, WqT);
  wconv_kernel<<<1024, 256, 0, stream>>>(Wk, WkT);
  wconv_kernel<<<1024, 256, 0, stream>>>(Wv, WvT);
  wconv_kernel<<<1024, 256, 0, stream>>>(Wo, WoT);
  ln_kernel<<<BBb * NNN, 256, 0, stream>>>(X, lng, lnb, Xn);
  gemm_kernel<<<(2048 / 128) * 8, 256, 0, stream>>>(Xn, WqT, bq, Qb, 2048, 0);
  gemm_kernel<<<(2048 / 128) * 8, 256, 0, stream>>>(Xn, WkT, bk, Kb, 2048, 0);
  gemm_kernel<<<(2048 / 128) * 8, 256, 0, stream>>>(Xn, WvT, bv, Vb, 2048, 2);
  attn_kernel<<<BBb * HH * GG * 4, 256, 0, stream>>>(Qb, Kb, Vb, sd, ga, rho, Ob);
  gemm_kernel<<<(16384 / 128) * 8, 256, 0, stream>>>(Ob, WoT, bo, d_out, 16384, 1);
}
